// GATModel_32074815767392
// MI455X (gfx1250) — matrix operand order repaired and checked by reading, not compiled, not run on hardware
//
#include <hip/hip_runtime.h>

// ---------------------------------------------------------------------------
// GAT x4 for MI455X (gfx1250, wave32).  GEMMs via v_wmma_f32_16x16x32_bf16
// (fp32 accumulate), softmax-attention via L2 atomics. All tensors fit in the
// 192MB L2, so edge gather/scatter runs at L2 bandwidth, not HBM.
// ---------------------------------------------------------------------------

typedef __attribute__((ext_vector_type(16))) __bf16 v16bf;
typedef __attribute__((ext_vector_type(8)))  float  v8f;

#define N_NODES 20000
#define LEAKY   0.2f

// ---------------- helpers ----------------
__device__ __forceinline__ unsigned fkey(float x) {
    unsigned b = __float_as_uint(x);
    return b ^ ((b & 0x80000000u) ? 0xFFFFFFFFu : 0x80000000u);
}
__device__ __forceinline__ float fkey_dec(unsigned k) {
    unsigned b = (k & 0x80000000u) ? (k ^ 0x80000000u) : ~k;
    return __uint_as_float(b);
}

// ---------------- elementwise / staging ----------------
__global__ void k_f32_to_bf16(const float* __restrict__ src,
                              __bf16* __restrict__ dst, int n) {
    int i = blockIdx.x * blockDim.x + threadIdx.x;
    if (i < n) dst[i] = (__bf16)src[i];
}

// Pack W (din x dout, row-major f32) into B-fragment layout:
// per (kt,nt) 32x16 tile: 32 lanes x 16 contiguous bf16 (lane n<16 -> col n,
// K=0..7,16..23 of tile; lane n>=16 -> col n-16, K=8..15,24..31).
__global__ void k_pack_w(const float* __restrict__ W, __bf16* __restrict__ wp,
                         int din, int dout) {
    int idx = blockIdx.x * blockDim.x + threadIdx.x;
    if (idx >= din * dout) return;
    int i    = idx & 15;
    int lane = (idx >> 4) & 31;
    int tile = idx >> 9;
    int ntN  = dout >> 4;
    int kt   = tile / ntN;
    int nt   = tile % ntN;
    int k = (kt << 5) + ((lane >> 4) << 3) + (i & 7) + ((i >> 3) << 4);
    int n = (nt << 4) + (lane & 15);
    wp[idx] = (__bf16)W[(size_t)k * dout + n];
}

__global__ void k_zero_f32(float* __restrict__ p, int n) {
    int i = blockIdx.x * blockDim.x + threadIdx.x;
    if (i < n) p[i] = 0.0f;
}

__global__ void k_init_nodes(unsigned* __restrict__ mkey, float* __restrict__ s, int n) {
    int i = blockIdx.x * blockDim.x + threadIdx.x;
    if (i < n) { mkey[i] = 0u; s[i] = 0.0f; }
}

// ---------------- WMMA GEMM: H = Xbf16 @ Wpack, fp32 accum ----------------
// One wave computes a 16x64 strip (4 n-tiles sharing the A fragment).
__global__ void k_gemm(const __bf16* __restrict__ xbf,
                       const __bf16* __restrict__ wp,
                       float* __restrict__ H, int M, int din, int dout) {
    int wave = (blockIdx.x * blockDim.x + threadIdx.x) >> 5;
    int lane = threadIdx.x & 31;
    int ngroups = dout >> 6;
    int mtile = wave / ngroups;
    int ng    = wave % ngroups;
    if (mtile >= (M >> 4)) return;
    int m0   = mtile << 4;
    int half = lane >> 4;       // ISA 16-bit A layout: lanes 16..31 hold K+8/K+24
    int l15  = lane & 15;
    int ntN  = dout >> 4;
    int nt0  = ng << 2;

    v8f c[4] = {};
    const __bf16* arow = xbf + (size_t)(m0 + l15) * din + half * 8;
    int kTiles = din >> 5;
    for (int kt = 0; kt < kTiles; ++kt) {
        union { v16bf v; uint4 u[2]; } a;
        const __bf16* ap = arow + (kt << 5);
        a.u[0] = *(const uint4*)(ap);        // K +0..7  (or +8..15)
        a.u[1] = *(const uint4*)(ap + 16);   // K +16..23 (or +24..31)
#pragma unroll
        for (int t = 0; t < 4; ++t) {
            union { v16bf v; uint4 u[2]; } b;
            const __bf16* bp = wp + ((((size_t)kt * ntN) + nt0 + t) * 32 + lane) * 16;
            b.u[0] = *(const uint4*)(bp);
            b.u[1] = *(const uint4*)(bp + 8);
            c[t] = __builtin_amdgcn_wmma_f32_16x16x32_bf16(
                       false, a.v, false, b.v, (short)0, c[t], false, false);
        }
    }
#pragma unroll
    for (int t = 0; t < 4; ++t) {
        int n = ((nt0 + t) << 4) + l15;
        float* out = H + (size_t)(m0 + half * 8) * dout + n;
#pragma unroll
        for (int j = 0; j < 8; ++j)
            out[(size_t)j * dout] = c[t][j];   // row m0 + half*8 + j
    }
}

// ---------------- attention scalars: el = h.al, er = h.ar (wave/node) -----
__global__ void k_eler(const float* __restrict__ H,
                       const float* __restrict__ al, const float* __restrict__ ar,
                       float* __restrict__ el, float* __restrict__ er,
                       int N, int dout) {
    int wave = (blockIdx.x * blockDim.x + threadIdx.x) >> 5;
    int lane = threadIdx.x & 31;
    if (wave >= N) return;
    const float* h = H + (size_t)wave * dout;
    float a = 0.f, b = 0.f;
    for (int j = lane; j < dout; j += 32) {
        float v = h[j];
        a += v * al[j];
        b += v * ar[j];
    }
#pragma unroll
    for (int off = 16; off; off >>= 1) {
        a += __shfl_down(a, off, 32);
        b += __shfl_down(b, off, 32);
    }
    if (lane == 0) { el[wave] = a; er[wave] = b; }
}

// ---------------- edge pass 1: leaky-relu score + segment max -------------
__global__ void k_edge_max(const float* __restrict__ el, const float* __restrict__ er,
                           const int* __restrict__ src, const int* __restrict__ dst,
                           float* __restrict__ score, unsigned* __restrict__ mkey, int E) {
    int e = blockIdx.x * blockDim.x + threadIdx.x;
    if (e >= E) return;
    float v = el[src[e]] + er[dst[e]];
    v = (v > 0.f) ? v : LEAKY * v;
    score[e] = v;
    atomicMax(&mkey[dst[e]], fkey(v));
}

__global__ void k_decode_m(const unsigned* __restrict__ mkey, float* __restrict__ m, int n) {
    int i = blockIdx.x * blockDim.x + threadIdx.x;
    if (i < n) m[i] = fkey_dec(mkey[i]);
}

// ---------------- edge pass 2: exp + segment sum --------------------------
__global__ void k_edge_exp(const float* __restrict__ score, const float* __restrict__ m,
                           const int* __restrict__ dst,
                           float* __restrict__ s, float* __restrict__ ex, int E) {
    int e = blockIdx.x * blockDim.x + threadIdx.x;
    if (e >= E) return;
    int d = dst[e];
    float v = __expf(score[e] - m[d]);
    ex[e] = v;
    unsafeAtomicAdd(&s[d], v);
}

// ---------------- edge pass 3: out[dst] += alpha * h[src] -----------------
__global__ void k_aggregate(const float* __restrict__ H,
                            const float* __restrict__ ex, const float* __restrict__ s,
                            const int* __restrict__ src, const int* __restrict__ dst,
                            float* __restrict__ A, int E, int dout) {
    long long idx = (long long)blockIdx.x * blockDim.x + threadIdx.x;
    int chunks = dout >> 3;
    long long total = (long long)E * chunks;
    if (idx >= total) return;
    int e  = (int)(idx / chunks);
    int f0 = ((int)(idx % chunks)) << 3;
    int sn = src[e], dn = dst[e];
    float alpha = ex[e] / s[dn];
    const float* hp = H + (size_t)sn * dout + f0;
    float* ap = A + (size_t)dn * dout + f0;
#pragma unroll
    for (int j = 0; j < 8; ++j)
        unsafeAtomicAdd(ap + j, hp[j] * alpha);
}

// ---------------- epilogue: tanh(out + b) ---------------------------------
__global__ void k_bias_tanh(float* __restrict__ A, const float* __restrict__ b,
                            int N, int dout) {
    long long idx = (long long)blockIdx.x * blockDim.x + threadIdx.x;
    if (idx >= (long long)N * dout) return;
    int j = (int)(idx % dout);
    A[idx] = tanhf(A[idx] + b[j]);
}

// ---------------- mean pool over first (order+1) rows ---------------------
__global__ void k_pool(const float* __restrict__ A, float* __restrict__ pool,
                       const int* __restrict__ order) {
    int j = blockIdx.x * blockDim.x + threadIdx.x;
    if (j >= 1024) return;
    int rows = order[0] + 1;
    float acc = 0.f;
    for (int n = 0; n < rows; ++n) acc += A[(size_t)n * 1024 + j];
    pool[j] = acc / (float)rows;
}

// ---------------- logits + nonzero(rel) gather ----------------------------
__global__ void k_logits(const float* __restrict__ pool,
                         const float* __restrict__ relW, const float* __restrict__ relB,
                         const int* __restrict__ rel, float* __restrict__ out) {
    __shared__ float lg[64];
    int j = threadIdx.x;  // 64 threads
    float acc = relB[j];
    for (int i = 0; i < 1024; ++i) acc += pool[i] * relW[(size_t)i * 64 + j];
    lg[j] = acc;
    __syncthreads();
    if (j == 0) {
        int cnt = 0;
        for (int k = 0; k < 64; ++k) if (rel[k] != 0) out[cnt++] = lg[k];
        for (; cnt < 64; ++cnt) out[cnt] = lg[0];  // jnp.nonzero pads with idx 0
    }
}

// ---------------------------------------------------------------------------
extern "C" void kernel_launch(void* const* d_in, const int* in_sizes, int n_in,
                              void* d_out, int out_size, void* d_ws, size_t ws_size,
                              hipStream_t stream) {
    const float* feat = (const float*)d_in[0];
    const float* Wl[4]  = {(const float*)d_in[1], (const float*)d_in[5],
                           (const float*)d_in[9], (const float*)d_in[13]};
    const float* all[4] = {(const float*)d_in[2], (const float*)d_in[6],
                           (const float*)d_in[10], (const float*)d_in[14]};
    const float* arl[4] = {(const float*)d_in[3], (const float*)d_in[7],
                           (const float*)d_in[11], (const float*)d_in[15]};
    const float* bl[4]  = {(const float*)d_in[4], (const float*)d_in[8],
                           (const float*)d_in[12], (const float*)d_in[16]};
    const float* relW = (const float*)d_in[17];
    const float* relB = (const float*)d_in[18];
    const int*   src  = (const int*)d_in[19];
    const int*   dst  = (const int*)d_in[20];
    const int*   rel  = (const int*)d_in[21];
    const int*   ordp = (const int*)d_in[22];
    const int E = in_sizes[19];
    const int N = N_NODES;
    const int dims[5] = {64, 128, 256, 512, 1024};

    // ---- carve workspace ----
    char* ws = (char*)d_ws;
    size_t off = 0;
    auto carve = [&](size_t bytes) {
        void* p = ws + off;
        off += (bytes + 255) & ~(size_t)255;
        return p;
    };
    float*    A     = (float*)   carve((size_t)N * 1024 * 4);  // agg / next-layer x
    float*    H     = (float*)   carve((size_t)N * 1024 * 4);  // h = x @ W
    __bf16*   xbf   = (__bf16*)  carve((size_t)N * 512 * 2);
    __bf16*   wpack = (__bf16*)  carve((size_t)512 * 1024 * 2);
    float*    el    = (float*)   carve((size_t)N * 4);
    float*    er    = (float*)   carve((size_t)N * 4);
    unsigned* mkey  = (unsigned*)carve((size_t)N * 4);
    float*    mval  = (float*)   carve((size_t)N * 4);
    float*    ssum  = (float*)   carve((size_t)N * 4);
    float*    score = (float*)   carve((size_t)E * 4);
    float*    exv   = (float*)   carve((size_t)E * 4);
    float*    pool  = (float*)   carve(1024 * 4);

    const int T = 256;
    auto blk = [](long long n, int t) { return (unsigned)((n + t - 1) / t); };

    const float* x = feat;
    for (int L = 0; L < 4; ++L) {
        const int din = dims[L], dout = dims[L + 1];

        // stage x -> bf16 (x is no longer needed afterwards; A can be reused)
        k_f32_to_bf16<<<blk((long long)N * din, T), T, 0, stream>>>(x, xbf, N * din);
        // pack W into B-fragment layout
        k_pack_w<<<blk((long long)din * dout, T), T, 0, stream>>>(Wl[L], wpack, din, dout);
        // H = x @ W via v_wmma_f32_16x16x32_bf16
        long long waves = (long long)(N / 16) * (dout / 64);
        k_gemm<<<blk(waves * 32, T), T, 0, stream>>>(xbf, wpack, H, N, din, dout);
        // el/er
        k_eler<<<blk((long long)N * 32, T), T, 0, stream>>>(H, all[L], arl[L], el, er, N, dout);
        // zero aggregation target + per-node softmax state
        k_zero_f32<<<blk((long long)N * dout, T), T, 0, stream>>>(A, N * dout);
        k_init_nodes<<<blk(N, T), T, 0, stream>>>(mkey, ssum, N);
        // softmax over incoming edges
        k_edge_max<<<blk(E, T), T, 0, stream>>>(el, er, src, dst, score, mkey, E);
        k_decode_m<<<blk(N, T), T, 0, stream>>>(mkey, mval, N);
        k_edge_exp<<<blk(E, T), T, 0, stream>>>(score, mval, dst, ssum, exv, E);
        // weighted aggregation
        long long tot = (long long)E * (dout / 8);
        k_aggregate<<<blk(tot, T), T, 0, stream>>>(H, exv, ssum, src, dst, A, E, dout);
        // activation
        k_bias_tanh<<<blk((long long)N * dout, T), T, 0, stream>>>(A, bl[L], N, dout);
        x = A;
    }

    k_pool<<<4, 256, 0, stream>>>(A, pool, ordp);
    k_logits<<<1, 64, 0, stream>>>(pool, relW, relB, rel, (float*)d_out);
}
